// SCC_45303315038762
// MI455X (gfx1250) — matrix operand order repaired and checked by reading, not compiled, not run on hardware
//
#include <hip/hip_runtime.h>

// ---------------------------------------------------------------------------
// CDNA5 / gfx1250 SCC block. All dense GEMMs via v_wmma_f32_16x16x32_bf16.
// All WMMA operands are pre-staged as zero-padded bf16 so fragment loads are
// unconditional 16-byte vector loads (no divergent guards, no per-load waits).
// ---------------------------------------------------------------------------

typedef __attribute__((ext_vector_type(16))) __bf16 v16bf;
typedef __attribute__((ext_vector_type(8)))  __bf16 v8bf;
typedef __attribute__((ext_vector_type(8)))  float  v8f;

#define DIMC 180
#define C5   36
#define NHH  6
#define HD   15
#define LW   256     // tokens per 16x16 window
#define MLW  64      // downsampled tokens per window
#define BWIN 512     // number of windows (2 * 16 * 16)
#define NPIX 131072  // 2 * 256 * 256
#define IMG  256

// padded leading dims
#define XP   192     // x / dfe_lin / cat / proj K-pad (180 -> 192)
#define T2P  64      // conv3 K-pad (36 -> 64)
#define QTP  192     // qtok row: 6 heads * 32 (head_dim 15 padded to 32)
#define VTP  96      // vtok row: 6 heads * 16
#define CMP  96      // channel-major rows (padded channels)
#define VPP  32      // vp row pad (15 -> 32)

__device__ __forceinline__ v8f wmma_bf16(v16bf a, v16bf b, v8f c) {
  return __builtin_amdgcn_wmma_f32_16x16x32_bf16(false, a, false, b, (short)0, c,
                                                 false, false);
}

__device__ __forceinline__ v16bf frag_cat(v8bf lo, v8bf hi) {
  return __builtin_shufflevector(lo, hi, 0, 1, 2, 3, 4, 5, 6, 7, 8, 9, 10, 11, 12,
                                 13, 14, 15);
}

// A fragment (16x32 bf16 layout): lane-half hl; element e -> K=k0+hl*16+e.
// Two contiguous 16B loads from the lane's row.
__device__ __forceinline__ v16bf load_a_bf(const __bf16* row, int k0, int hl) {
  v8bf lo = *(const v8bf*)(row + k0 + hl * 16);
  v8bf hi = *(const v8bf*)(row + k0 + hl * 16 + 8);
  return frag_cat(lo, hi);
}

// B fragment (32x16 bf16 layout): element e -> K = k0 + hl*16 + e; the lane's
// "row" is the N-major (transposed) row for column n. Two 16B loads.
__device__ __forceinline__ v16bf load_b_bf(const __bf16* row, int k0, int hl) {
  v8bf lo = *(const v8bf*)(row + k0 + hl * 16);
  v8bf hi = *(const v8bf*)(row + k0 + hl * 16 + 8);
  return frag_cat(lo, hi);
}

// ---------------------------------------------------------------------------
// zero-fill (16B granules), grid-stride
// ---------------------------------------------------------------------------
__global__ void k_zero(uint4* __restrict__ p, long long n16) {
  uint4 z = {0u, 0u, 0u, 0u};
  long long stride = (long long)gridDim.x * blockDim.x;
  for (long long i = blockIdx.x * (long long)blockDim.x + threadIdx.x; i < n16;
       i += stride)
    p[i] = z;
}

// transpose + pad weights: src (K x N) f32 -> dst (Np x Kp) bf16, dst[n][k]
__global__ void k_wtrans(const float* __restrict__ src, __bf16* __restrict__ dst,
                         int K, int N, int Kp, int Np) {
  int i = blockIdx.x * blockDim.x + threadIdx.x;
  if (i >= Kp * Np) return;
  int n = i / Kp, k = i % Kp;
  dst[i] = (k < K && n < N) ? (__bf16)src[k * N + n] : (__bf16)0.0f;
}

// x (NPIX x 180 f32) -> xb (NPIX x 192 bf16, zero-padded)
__global__ void k_prep_x(const float* __restrict__ x, __bf16* __restrict__ xb) {
  int row = blockIdx.x * blockDim.x + threadIdx.x;
  if (row >= NPIX) return;
  const float* s = x + row * DIMC;
  __bf16* d = xb + (size_t)row * XP;
  for (int c = 0; c < DIMC; ++c) d[c] = (__bf16)s[c];
  for (int c = DIMC; c < XP; ++c) d[c] = (__bf16)0.0f;
}

// ---------------------------------------------------------------------------
// Kernel 1: t1 = leaky_relu(x @ conv1_w + conv1_b) -> (NPIX x 36) f32
// ---------------------------------------------------------------------------
__global__ void k_dfe1(const __bf16* __restrict__ xb, const __bf16* __restrict__ w1t,
                       const float* __restrict__ b1, float* __restrict__ t1) {
  int wave = blockIdx.x * (blockDim.x >> 5) + (threadIdx.x >> 5);
  int lane = threadIdx.x & 31;
  int nn = lane & 15, hl = lane >> 4;
  int mt = wave / 3, nt = wave % 3;
  int m0 = mt * 16, n0 = nt * 16;
  const __bf16* arow = xb + (size_t)(m0 + nn) * XP;
  const __bf16* brow = w1t + (n0 + nn) * XP;
  v8f acc = {0.f, 0.f, 0.f, 0.f, 0.f, 0.f, 0.f, 0.f};
#pragma unroll
  for (int k0 = 0; k0 < XP; k0 += 32)
    acc = wmma_bf16(load_a_bf(arow, k0, hl), load_b_bf(brow, k0, hl), acc);
  int n = n0 + nn;
  if (n < C5) {
    float bias = b1[n];
#pragma unroll
    for (int i = 0; i < 8; ++i) {
      int row = m0 + i + hl * 8;
      float v = acc[i] + bias;
      t1[row * C5 + n] = (v > 0.f) ? v : 0.2f * v;
    }
  }
}

// ---------------------------------------------------------------------------
// Kernel 2: t2b = leaky_relu(conv3x3(t1) + b) -> (NPIX x 64) bf16 (pad zeroed)
// ---------------------------------------------------------------------------
__global__ void k_conv3(const float* __restrict__ t1, const float* __restrict__ w2,
                        const float* __restrict__ cb, __bf16* __restrict__ t2b) {
  __shared__ float patch[18 * 18 * 36];  // 45.6 KB
  int bid = blockIdx.x;
  int bb = bid >> 8, th = (bid >> 4) & 15, tw = bid & 15;
  int h0 = th * 16, w0 = tw * 16;
  int tid = threadIdx.x;
  __builtin_prefetch(w2, 0, 3);
  for (int idx = tid; idx < 18 * 18 * 36; idx += 256) {
    int c = idx % 36, rest = idx / 36;
    int pw_ = rest % 18, ph = rest / 18;
    int gh = h0 + ph - 1, gw = w0 + pw_ - 1;
    float v = 0.f;
    if (gh >= 0 && gh < IMG && gw >= 0 && gw < IMG)
      v = t1[((bb * IMG + gh) * IMG + gw) * C5 + c];
    patch[idx] = v;
  }
  __syncthreads();
  int i = tid >> 4, j = tid & 15;
  float acc[C5];
#pragma unroll
  for (int co = 0; co < C5; ++co) acc[co] = cb[co];
  for (int dy = 0; dy < 3; ++dy)
    for (int dx = 0; dx < 3; ++dx) {
      const float* wt = w2 + (dy * 3 + dx) * C5 * C5;
      const float* pr = patch + ((i + dy) * 18 + (j + dx)) * C5;
      for (int ci = 0; ci < C5; ++ci) {
        float a = pr[ci];
        const float* wr = wt + ci * C5;
#pragma unroll
        for (int co = 0; co < C5; ++co) acc[co] += a * wr[co];
      }
    }
  int pix = (bb * IMG + h0 + i) * IMG + (w0 + j);
  __bf16* d = t2b + (size_t)pix * T2P;
#pragma unroll
  for (int co = 0; co < C5; ++co) {
    float v = acc[co];
    d[co] = (__bf16)((v > 0.f) ? v : 0.2f * v);
  }
#pragma unroll
  for (int co = C5; co < T2P; ++co) d[co] = (__bf16)0.0f;
}

// ---------------------------------------------------------------------------
// Kernel 3 (fused DFE tail + q/v split):
//   qv = (t2 @ conv3_w + b3) * (x @ dfe_lin_w + dfe_lin_b)
// Writes the q half into qtok (token-major, 32/head) + qwin (channel-major),
// the v half into vtok (token-major, 16/head) + vwin (channel-major), bf16.
// ---------------------------------------------------------------------------
__global__ void k_dfe3(const __bf16* __restrict__ t2b, const __bf16* __restrict__ w3t,
                       const float* __restrict__ b3, const __bf16* __restrict__ xb,
                       const __bf16* __restrict__ wlt, const float* __restrict__ bl,
                       __bf16* __restrict__ qtok, __bf16* __restrict__ vtok,
                       __bf16* __restrict__ qwin, __bf16* __restrict__ vwin) {
  int wave = blockIdx.x * (blockDim.x >> 5) + (threadIdx.x >> 5);
  int lane = threadIdx.x & 31;
  int nn = lane & 15, hl = lane >> 4;
  int mt = wave / 12, nt = wave % 12;
  int m0 = mt * 16, n0 = nt * 16;
  v8f acc1 = {0.f, 0.f, 0.f, 0.f, 0.f, 0.f, 0.f, 0.f};
  v8f acc2 = acc1;
  {  // y = t2 @ conv3_w   (K = 36, padded 64)
    const __bf16* arow = t2b + (size_t)(m0 + nn) * T2P;
    const __bf16* brow = w3t + (n0 + nn) * T2P;
#pragma unroll
    for (int k0 = 0; k0 < T2P; k0 += 32)
      acc1 = wmma_bf16(load_a_bf(arow, k0, hl), load_b_bf(brow, k0, hl), acc1);
  }
  {  // lin = x @ dfe_lin_w (K = 180, padded 192)
    const __bf16* arow = xb + (size_t)(m0 + nn) * XP;
    const __bf16* brow = wlt + (n0 + nn) * XP;
#pragma unroll
    for (int k0 = 0; k0 < XP; k0 += 32)
      acc2 = wmma_bf16(load_a_bf(arow, k0, hl), load_b_bf(brow, k0, hl), acc2);
  }
  int n = n0 + nn;
  if (n < DIMC) {
    float bb3 = b3[n], bbl = bl[n];
    bool isq = n < (NHH * HD);
    int cidx = isq ? n : n - NHH * HD;
    int h = cidx / HD, d = cidx % HD;
    int tok_off = isq ? (h * 32 + d) : (h * 16 + d);
    int cm_row = h * 16 + d;  // padded channel index (0..95)
#pragma unroll
    for (int i = 0; i < 8; ++i) {
      int pix = m0 + i + hl * 8;
      float v = (acc1[i] + bb3) * (acc2[i] + bbl);
      __bf16 bv = (__bf16)v;
      int b_ = pix >> 16, gh = (pix >> 8) & 255, gw = pix & 255;
      int w = (b_ << 8) | ((gh >> 4) << 4) | (gw >> 4);
      int l = ((gh & 15) << 4) | (gw & 15);
      if (isq) {
        qtok[(size_t)(w * LW + l) * QTP + tok_off] = bv;
        qwin[(size_t)(w * CMP + cm_row) * LW + l] = bv;
      } else {
        vtok[(size_t)(w * LW + l) * VTP + tok_off] = bv;
        vwin[(size_t)(w * CMP + cm_row) * LW + l] = bv;
      }
    }
  }
}

// ---------------------------------------------------------------------------
// Kernel 4: vp[w,h,m,c] = sl_b + sum_r sl_w[r] * v[token(m,r), c], bf16,
// row padded to 32 (cols 15..31 zero).
// ---------------------------------------------------------------------------
__global__ void k_vp(const __bf16* __restrict__ vtok, const float* __restrict__ slw,
                     const float* __restrict__ slb, __bf16* __restrict__ vpb) {
  int t = blockIdx.x * blockDim.x + threadIdx.x;
  if (t >= BWIN * NHH * MLW) return;
  int w = t / (NHH * MLW);
  int rem = t % (NHH * MLW);
  int h = rem / MLW, m = rem % MLW;
  int mh = m >> 3, mw = m & 7;
  const __bf16* base = vtok + (size_t)w * LW * VTP + h * 16;
  int l0 = (mh * 2) * 16 + (mw * 2);
  float s0 = slw[0], s1 = slw[1], s2 = slw[2], s3 = slw[3], bb = slb[0];
  __bf16* dst = vpb + (size_t)t * VPP;
  for (int c = 0; c < HD; ++c) {
    float s = bb + s0 * (float)base[(l0)*VTP + c] + s1 * (float)base[(l0 + 1) * VTP + c] +
              s2 * (float)base[(l0 + 16) * VTP + c] +
              s3 * (float)base[(l0 + 17) * VTP + c];
    dst[c] = (__bf16)s;
  }
  for (int c = HD; c < VPP; ++c) dst[c] = (__bf16)0.0f;
}

// ---------------------------------------------------------------------------
// Kernel 5a: DynamicPosBias MLP -> p table (961 x 6)
// ---------------------------------------------------------------------------
__device__ __forceinline__ void ln_relu11(float* a, const float* __restrict__ g,
                                          const float* __restrict__ b) {
  float mn = 0.f;
#pragma unroll
  for (int i = 0; i < 11; ++i) mn += a[i];
  mn *= (1.0f / 11.0f);
  float var = 0.f;
#pragma unroll
  for (int i = 0; i < 11; ++i) { float d = a[i] - mn; var += d * d; }
  var *= (1.0f / 11.0f);
  float inv = rsqrtf(var + 1e-5f);
#pragma unroll
  for (int i = 0; i < 11; ++i) {
    float v = (a[i] - mn) * inv * g[i] + b[i];
    a[i] = v > 0.f ? v : 0.f;
  }
}

__global__ void k_posmlp(const float* __restrict__ pw, const float* __restrict__ pb,
                         const float* __restrict__ g1, const float* __restrict__ b1,
                         const float* __restrict__ m1w, const float* __restrict__ m1b,
                         const float* __restrict__ g2, const float* __restrict__ b2,
                         const float* __restrict__ m2w, const float* __restrict__ m2b,
                         const float* __restrict__ g3, const float* __restrict__ b3,
                         const float* __restrict__ m3w, const float* __restrict__ m3b,
                         float* __restrict__ ptab) {
  int g = blockIdx.x * blockDim.x + threadIdx.x;
  if (g >= 961) return;
  float dh = (float)(g / 31 - 15), dw = (float)(g % 31 - 15);
  float a[11], t[11];
#pragma unroll
  for (int j = 0; j < 11; ++j) a[j] = dh * pw[j] + dw * pw[11 + j] + pb[j];
  ln_relu11(a, g1, b1);
#pragma unroll
  for (int j = 0; j < 11; ++j) {
    float s = m1b[j];
#pragma unroll
    for (int i = 0; i < 11; ++i) s += a[i] * m1w[i * 11 + j];
    t[j] = s;
  }
#pragma unroll
  for (int j = 0; j < 11; ++j) a[j] = t[j];
  ln_relu11(a, g2, b2);
#pragma unroll
  for (int j = 0; j < 11; ++j) {
    float s = m2b[j];
#pragma unroll
    for (int i = 0; i < 11; ++i) s += a[i] * m2w[i * 11 + j];
    t[j] = s;
  }
#pragma unroll
  for (int j = 0; j < 11; ++j) a[j] = t[j];
  ln_relu11(a, g3, b3);
#pragma unroll
  for (int h = 0; h < NHH; ++h) {
    float s = m3b[h];
#pragma unroll
    for (int i = 0; i < 11; ++i) s += a[i] * m3w[i * NHH + h];
    ptab[g * NHH + h] = s;
  }
}

// ---------------------------------------------------------------------------
// Kernel 5b: rpb[h, l, m]
// ---------------------------------------------------------------------------
__global__ void k_rpb(const float* __restrict__ ptab, float* __restrict__ rpb) {
  int t = blockIdx.x * blockDim.x + threadIdx.x;
  int l = t >> 6, m = t & 63;
  int i1 = l >> 4, j1 = l & 15;
  int mh = m >> 3, mw = m & 7;
  float acc[NHH] = {0.f, 0.f, 0.f, 0.f, 0.f, 0.f};
#pragma unroll
  for (int rh = 0; rh < 2; ++rh)
#pragma unroll
    for (int rw = 0; rw < 2; ++rw) {
      int i2 = mh * 2 + rh, j2 = mw * 2 + rw;
      int idx = (i1 - i2 + 15) * 31 + (j1 - j2 + 15);
      const float* pr = ptab + idx * NHH;
#pragma unroll
      for (int h = 0; h < NHH; ++h) acc[h] += pr[h];
    }
#pragma unroll
  for (int h = 0; h < NHH; ++h) rpb[(h * LW + l) * MLW + m] = acc[h] * 0.25f;
}

// ---------------------------------------------------------------------------
// Kernel 6: spatial branch. One block = one (window, head).
// ---------------------------------------------------------------------------
__global__ void k_spatial(const __bf16* __restrict__ qtok,
                          const __bf16* __restrict__ vpb,
                          const float* __restrict__ rpb, __bf16* __restrict__ catb) {
  int w = blockIdx.x / NHH, h = blockIdx.x % NHH;
  __shared__ __align__(16) __bf16 vp_sh[MLW * VPP];       // 4 KB (rows 32-wide)
  __shared__ __align__(16) __bf16 corr_sh[4][16 * MLW];   // 8 KB
  const __bf16* vsrc = vpb + (size_t)(w * NHH + h) * MLW * VPP;
  for (int i = threadIdx.x; i < MLW * VPP; i += blockDim.x) vp_sh[i] = vsrc[i];
  __syncthreads();

  int wv = threadIdx.x >> 5, lane = threadIdx.x & 31;
  int m = lane & 15, hl = lane >> 4;
  int wb = (((w >> 8) * IMG + ((w >> 4) & 15) * 16) * IMG) + (w & 15) * 16;
  v8f zero = {0.f, 0.f, 0.f, 0.f, 0.f, 0.f, 0.f, 0.f};

  for (int sub = 0; sub < 4; ++sub) {
    int l0 = (wv * 4 + sub) * 16;
    // A = q tile (16 x 15, padded to 32): contiguous per-head slice of qtok
    const __bf16* qrow = qtok + (size_t)(w * LW + l0 + m) * QTP + h * 32;
    v16bf aq = load_a_bf(qrow, 0, hl);
    // corr = q @ vp^T (4 N-tiles of downsampled tokens)
    v8f acc[4];
#pragma unroll
    for (int nt = 0; nt < 4; ++nt) {
      v16bf b = load_b_bf(vp_sh + (nt * 16 + m) * VPP, 0, hl);
      acc[nt] = wmma_bf16(aq, b, zero);
    }
    // scale + dynamic position bias; stage corr as bf16 A-operand
#pragma unroll
    for (int nt = 0; nt < 4; ++nt) {
      int mc = nt * 16 + m;
#pragma unroll
      for (int i = 0; i < 8; ++i) {
        int l = l0 + i + hl * 8;
        float v = acc[nt][i] * (1.0f / 15.0f) + rpb[(h * LW + l) * MLW + mc];
        corr_sh[wv][(i + hl * 8) * MLW + mc] = (__bf16)v;
      }
    }
    // x_sp = corr @ vp  (K = 64, two chunks)
    v8f o = zero;
#pragma unroll
    for (int kc = 0; kc < 2; ++kc) {
      int k0 = kc * 32;
      v16bf a2 = load_a_bf(corr_sh[wv] + m * MLW, k0, hl);
      v16bf b2;
#pragma unroll
      for (int e = 0; e < 16; ++e) b2[e] = vp_sh[(k0 + hl * 16 + e) * VPP + m];
      o = wmma_bf16(a2, b2, o);
    }
    if (m < HD) {
#pragma unroll
      for (int i = 0; i < 8; ++i) {
        int l = l0 + i + hl * 8;
        int pix = wb + (l >> 4) * IMG + (l & 15);
        catb[(size_t)pix * XP + h * HD + m] = (__bf16)o[i];
      }
    }
  }
}

// ---------------------------------------------------------------------------
// Kernel 7: channel branch. One block = one window.
// ---------------------------------------------------------------------------
__global__ void k_channel(const __bf16* __restrict__ qwin,
                          const __bf16* __restrict__ vwin,
                          const __bf16* __restrict__ vtok,
                          __bf16* __restrict__ catb) {
  int w = blockIdx.x;
  __shared__ __align__(16) __bf16 cc_sh[96 * 96];  // 18 KB
  int wv = threadIdx.x >> 5, lane = threadIdx.x & 31;
  int nn = lane & 15, hl = lane >> 4;
  int wb = (((w >> 8) * IMG + ((w >> 4) & 15) * 16) * IMG) + (w & 15) * 16;
  v8f zero = {0.f, 0.f, 0.f, 0.f, 0.f, 0.f, 0.f, 0.f};

  // phase 1: cc[c, d] = sum_l qc[l, c] * vc[l, d] / 256  (channel-major inputs)
  for (int t = wv; t < 36; t += 4) {
    int mt = t / 6, nt = t % 6;
    const __bf16* arow = qwin + (size_t)(w * CMP + mt * 16 + nn) * LW;
    const __bf16* brow = vwin + (size_t)(w * CMP + nt * 16 + nn) * LW;
    v8f acc = zero;
#pragma unroll
    for (int k0 = 0; k0 < LW; k0 += 32)
      acc = wmma_bf16(load_a_bf(arow, k0, hl), load_b_bf(brow, k0, hl), acc);
#pragma unroll
    for (int i = 0; i < 8; ++i) {
      int row = mt * 16 + i + hl * 8;
      cc_sh[row * 96 + nt * 16 + nn] = (__bf16)(acc[i] * (1.0f / 256.0f));
    }
  }
  __syncthreads();

  // phase 2: x_ch[l, c] = sum_d cc[c, d] * vc[l, d]  (K = 96 padded channels)
  for (int t = wv; t < 96; t += 4) {
    int mt = t / 6, nt = t % 6;
    int l0 = mt * 16;
    int c = nt * 16 + nn;
    int hc = c >> 4, dc = c & 15;
    const __bf16* arow = vtok + (size_t)(w * LW + l0 + nn) * VTP;
    const __bf16* brow = cc_sh + c * 96;
    v8f acc = zero;
#pragma unroll
    for (int kc = 0; kc < 3; ++kc) {
      int k0 = kc * 32;
      acc = wmma_bf16(load_a_bf(arow, k0, hl), load_b_bf(brow, k0, hl), acc);
    }
    if (dc < HD) {
#pragma unroll
      for (int i = 0; i < 8; ++i) {
        int l = l0 + i + hl * 8;
        int pix = wb + (l >> 4) * IMG + (l & 15);
        catb[(size_t)pix * XP + NHH * HD + hc * HD + dc] = (__bf16)acc[i];
      }
    }
  }
}

// ---------------------------------------------------------------------------
// Kernel 8: out = cat @ proj_w + proj_b  (K = 180 padded 192)
// ---------------------------------------------------------------------------
__global__ void k_proj(const __bf16* __restrict__ catb, const __bf16* __restrict__ pjt,
                       const float* __restrict__ pb, float* __restrict__ out) {
  int wave = blockIdx.x * (blockDim.x >> 5) + (threadIdx.x >> 5);
  int lane = threadIdx.x & 31;
  int nn = lane & 15, hl = lane >> 4;
  int mt = wave / 12, nt = wave % 12;
  int m0 = mt * 16, n0 = nt * 16;
  const __bf16* arow = catb + (size_t)(m0 + nn) * XP;
  const __bf16* brow = pjt + (n0 + nn) * XP;
  v8f acc = {0.f, 0.f, 0.f, 0.f, 0.f, 0.f, 0.f, 0.f};
#pragma unroll
  for (int k0 = 0; k0 < XP; k0 += 32)
    acc = wmma_bf16(load_a_bf(arow, k0, hl), load_b_bf(brow, k0, hl), acc);
  int n = n0 + nn;
  if (n < DIMC) {
    float bias = pb[n];
#pragma unroll
    for (int i = 0; i < 8; ++i) {
      int row = m0 + i + hl * 8;
      out[row * DIMC + n] = acc[i] + bias;
    }
  }
}

// ---------------------------------------------------------------------------
extern "C" void kernel_launch(void* const* d_in, const int* in_sizes, int n_in,
                              void* d_out, int out_size, void* d_ws, size_t ws_size,
                              hipStream_t stream) {
  const float* x   = (const float*)d_in[0];
  const float* c1w = (const float*)d_in[1];
  const float* c1b = (const float*)d_in[2];
  const float* c2w = (const float*)d_in[3];
  const float* c2b = (const float*)d_in[4];
  const float* c3w = (const float*)d_in[5];
  const float* c3b = (const float*)d_in[6];
  const float* dlw = (const float*)d_in[7];
  const float* dlb = (const float*)d_in[8];
  const float* slw = (const float*)d_in[9];
  const float* slb = (const float*)d_in[10];
  const float* ppw = (const float*)d_in[11];
  const float* ppb = (const float*)d_in[12];
  const float* g1  = (const float*)d_in[13];
  const float* b1  = (const float*)d_in[14];
  const float* m1w = (const float*)d_in[15];
  const float* m1b = (const float*)d_in[16];
  const float* g2  = (const float*)d_in[17];
  const float* b2  = (const float*)d_in[18];
  const float* m2w = (const float*)d_in[19];
  const float* m2b = (const float*)d_in[20];
  const float* g3  = (const float*)d_in[21];
  const float* b3  = (const float*)d_in[22];
  const float* m3w = (const float*)d_in[23];
  const float* m3b = (const float*)d_in[24];
  const float* pjw = (const float*)d_in[25];
  const float* pjb = (const float*)d_in[26];
  float* out = (float*)d_out;

  char* ws = (char*)d_ws;
  float*  t1   = (float*)ws;   ws += (size_t)NPIX * C5 * 4;            // 18.9 MB
  __bf16* t2b  = (__bf16*)ws;  ws += (size_t)NPIX * T2P * 2;           // 16.8 MB
  __bf16* xb   = (__bf16*)ws;  ws += (size_t)NPIX * XP * 2;            // 50.3 MB
  // ---- contiguous zero-filled region (pads must be deterministic zeros) ----
  char* zbase = ws;
  __bf16* qtok = (__bf16*)ws;  ws += (size_t)BWIN * LW * QTP * 2;      // 50.3 MB
  __bf16* vtok = (__bf16*)ws;  ws += (size_t)BWIN * LW * VTP * 2;      // 25.2 MB
  __bf16* qwin = (__bf16*)ws;  ws += (size_t)BWIN * CMP * LW * 2;      // 25.2 MB
  __bf16* vwin = (__bf16*)ws;  ws += (size_t)BWIN * CMP * LW * 2;      // 25.2 MB
  __bf16* catb = (__bf16*)ws;  ws += (size_t)NPIX * XP * 2;            // 50.3 MB
  long long zbytes = (long long)(ws - zbase);
  // --------------------------------------------------------------------------
  __bf16* vpb  = (__bf16*)ws;  ws += (size_t)BWIN * NHH * MLW * VPP * 2; // 12.6 MB
  __bf16* w1t  = (__bf16*)ws;  ws += (size_t)48 * XP * 2;
  __bf16* w3t  = (__bf16*)ws;  ws += (size_t)XP * T2P * 2;
  __bf16* wlt  = (__bf16*)ws;  ws += (size_t)XP * XP * 2;
  __bf16* pjt  = (__bf16*)ws;  ws += (size_t)XP * XP * 2;
  float*  ptab = (float*)ws;   ws += (size_t)961 * NHH * 4;
  float*  rpb  = (float*)ws;   ws += (size_t)NHH * LW * MLW * 4;

  // prep
  k_zero<<<4096, 256, 0, stream>>>((uint4*)zbase, zbytes / 16);
  k_wtrans<<<36, 256, 0, stream>>>(c1w, w1t, DIMC, C5, XP, 48);
  k_wtrans<<<48, 256, 0, stream>>>(c3w, w3t, C5, DIMC, T2P, XP);
  k_wtrans<<<144, 256, 0, stream>>>(dlw, wlt, DIMC, DIMC, XP, XP);
  k_wtrans<<<144, 256, 0, stream>>>(pjw, pjt, DIMC, DIMC, XP, XP);
  k_prep_x<<<512, 256, 0, stream>>>(x, xb);
  // DFE
  k_dfe1<<<3072, 256, 0, stream>>>(xb, w1t, c1b, t1);
  k_conv3<<<512, 256, 0, stream>>>(t1, c2w, c2b, t2b);
  k_dfe3<<<12288, 256, 0, stream>>>(t2b, w3t, c3b, xb, wlt, dlb, qtok, vtok, qwin, vwin);
  // attention prep
  k_vp<<<768, 256, 0, stream>>>(vtok, slw, slb, vpb);
  k_posmlp<<<4, 256, 0, stream>>>(ppw, ppb, g1, b1, m1w, m1b, g2, b2, m2w, m2b,
                                  g3, b3, m3w, m3b, ptab);
  k_rpb<<<64, 256, 0, stream>>>(ptab, rpb);
  // attention
  k_spatial<<<BWIN * NHH, 128, 0, stream>>>(qtok, vpb, rpb, catb);
  k_channel<<<BWIN, 128, 0, stream>>>(qwin, vwin, vtok, catb);
  // projection
  k_proj<<<12288, 256, 0, stream>>>(catb, pjt, pjb, out);
}
